// MultiHeadSelfAttention_8701603742005
// MI455X (gfx1250) — matrix operand order repaired and checked by reading, not compiled, not run on hardware
//
#include <hip/hip_runtime.h>
#include <hip/hip_bf16.h>

// MI455X / gfx1250 wave32 WMMA implementation of fused multi-head attention.
// B=2, S=2048, D=1024, H=16, DH=64.  All matmuls via v_wmma_f32_16x16x32_f16.

typedef __attribute__((ext_vector_type(16))) _Float16 v16h;
typedef __attribute__((ext_vector_type(8)))  _Float16 v8h;
typedef __attribute__((ext_vector_type(8)))  float    v8f;

#define WMMA_F16(a, b, c) \
  __builtin_amdgcn_wmma_f32_16x16x32_f16(false, (a), false, (b), (short)0, (c), false, false)

__device__ __forceinline__ v8f zero_v8f() {
  v8f z = {0.f, 0.f, 0.f, 0.f, 0.f, 0.f, 0.f, 0.f};
  return z;
}

// A-fragment (16x32 f16, MxK). Lane layout used here (contiguous k, matched by B):
//   lane: M = lane&15, g = lane>>4
//   halves 0..7  -> K = g*16 + i       (contiguous 16B)
//   halves 8..15 -> K = g*16 + 8 + i   (contiguous 16B)
// p points at (row M of this lane, k0).
__device__ __forceinline__ v16h load_afrag(const _Float16* __restrict__ p, int g) {
  const v8h lo = *(const v8h*)(p + g * 16);
  const v8h hi = *(const v8h*)(p + g * 16 + 8);
  v16h a;
#pragma unroll
  for (int i = 0; i < 8; ++i) { a[i] = lo[i]; a[8 + i] = hi[i]; }
  return a;
}

// ---------------------------------------------------------------------------
// Stage 0: precision / layout conversion kernels
// ---------------------------------------------------------------------------
__global__ void cast_f32_to_f16_kernel(const float* __restrict__ src,
                                       _Float16* __restrict__ dst, int n) {
  int i = blockIdx.x * blockDim.x + threadIdx.x;
  if (i < n) dst[i] = (_Float16)src[i];
}

// Wq[H,D,DH] f32 -> WT[H*DH, D] f16  (transposed so B-operand reads are contiguous)
__global__ void pack_w_qkv_kernel(const float* __restrict__ src,
                                  _Float16* __restrict__ dst) {
  int i = blockIdx.x * blockDim.x + threadIdx.x;   // over H*D*DH = 1048576
  int dh = i & 63;
  int d  = (i >> 6) & 1023;
  int h  = i >> 16;
  dst[(size_t)(h * 64 + dh) * 1024 + d] = (_Float16)src[i];
}

// Wo[K,N] f32 -> WoT[N,K] f16
__global__ void pack_wo_kernel(const float* __restrict__ src,
                               _Float16* __restrict__ dst) {
  int i = blockIdx.x * blockDim.x + threadIdx.x;   // dst index, over 1048576
  int k = i & 1023;
  int n = i >> 10;
  dst[i] = (_Float16)src[(size_t)k * 1024 + n];
}

// ---------------------------------------------------------------------------
// Stage 1/3: generic WMMA GEMM.  C[M,N] = A[M,K] * BT[N,K]^T + bias[N]
// One wave computes a 32x64 tile (2 M-tiles x 4 N-tiles, 8 accumulators):
// 8 WMMAs per 12 b128 loads per k-step, B reused across M-tiles, A across
// N-tiles.  256 thr = 8 waves/block.
// mode 0: store f16 row-major [M,N]
// mode 1: store f16 as V-transposed [B,H,DH,S] (row m -> (b,t), col n -> (h,dh))
// mode 2: store f32 row-major [M,N]  (final output)
// ---------------------------------------------------------------------------
__global__ __launch_bounds__(256) void gemm_wmma_kernel(
    const _Float16* __restrict__ A, const _Float16* __restrict__ BT,
    const float* __restrict__ bias, void* __restrict__ Cout,
    int M, int N, int K, int mode) {
  const int lane = threadIdx.x & 31;
  const int gw   = blockIdx.x * 8 + (threadIdx.x >> 5);
  const int tilesN = N >> 6;
  const int m0 = (gw / tilesN) * 32;
  const int n0 = (gw % tilesN) * 64;
  if (m0 >= M) return;                       // wave-uniform
  const int l15 = lane & 15;
  const int g   = lane >> 4;

  const _Float16* arow0 = A + (size_t)(m0 + l15) * K;
  const _Float16* arow1 = arow0 + (size_t)16 * K;
  // B-fragment (contiguous k, same order as load_afrag): lane holds column n=lane&15, halves 0..15 -> K = g*16+i
  const _Float16* brow0 = BT + (size_t)(n0 + l15) * K + g * 16;
  const _Float16* brow1 = brow0 + (size_t)16 * K;
  const _Float16* brow2 = brow0 + (size_t)32 * K;
  const _Float16* brow3 = brow0 + (size_t)48 * K;

  v8f acc[2][4];
#pragma unroll
  for (int i = 0; i < 2; ++i)
#pragma unroll
    for (int j = 0; j < 4; ++j) acc[i][j] = zero_v8f();

  for (int k0 = 0; k0 < K; k0 += 32) {
    // Prefetch one k-step ahead (lowers to global_prefetch_b8 on gfx1250).
    __builtin_prefetch(arow0 + k0 + 32, 0, 3);
    __builtin_prefetch(brow0 + k0 + 32, 0, 3);

    v16h a0 = load_afrag(arow0 + k0, g);
    v16h a1 = load_afrag(arow1 + k0, g);
    v16h b0 = *(const v16h*)(brow0 + k0);
    v16h b1 = *(const v16h*)(brow1 + k0);
    v16h b2 = *(const v16h*)(brow2 + k0);
    v16h b3 = *(const v16h*)(brow3 + k0);
    acc[0][0] = WMMA_F16(a0, b0, acc[0][0]);
    acc[0][1] = WMMA_F16(a0, b1, acc[0][1]);
    acc[0][2] = WMMA_F16(a0, b2, acc[0][2]);
    acc[0][3] = WMMA_F16(a0, b3, acc[0][3]);
    acc[1][0] = WMMA_F16(a1, b0, acc[1][0]);
    acc[1][1] = WMMA_F16(a1, b1, acc[1][1]);
    acc[1][2] = WMMA_F16(a1, b2, acc[1][2]);
    acc[1][3] = WMMA_F16(a1, b3, acc[1][3]);
  }

#pragma unroll
  for (int i = 0; i < 2; ++i) {
#pragma unroll
    for (int j = 0; j < 4; ++j) {
      const int ncol = n0 + j * 16 + l15;
      const float bv = bias[ncol];
#pragma unroll
      for (int r = 0; r < 8; ++r) {
        const int mr = m0 + i * 16 + r + g * 8;  // C layout: lanes16-31 = M+8
        const float val = acc[i][j][r] + bv;
        if (mode == 0) {
          ((_Float16*)Cout)[(size_t)mr * N + ncol] = (_Float16)val;
        } else if (mode == 1) {
          const int bb = mr >> 11;               // batch from global row
          ((_Float16*)Cout)[(size_t)((bb << 10) + ncol) * 2048 + (mr & 2047)] =
              (_Float16)val;
        } else {
          ((float*)Cout)[(size_t)mr * N + ncol] = val;
        }
      }
    }
  }
}

// ---------------------------------------------------------------------------
// Stage 2: flash attention.  One wave owns 16 queries of one (b,h).
// Computes S^T = K*Q^T so softmax stats are per-lane (+ one shfl_xor 16),
// then O^T = V^T * P^T with V pre-transposed in memory.
// ---------------------------------------------------------------------------
__global__ __launch_bounds__(256) void flash_attn_kernel(
    const _Float16* __restrict__ Q,   // [B,S,H*DH] f16
    const _Float16* __restrict__ Km,  // [B,S,H*DH] f16
    const _Float16* __restrict__ Vt,  // [B,H,DH,S] f16
    _Float16* __restrict__ O) {       // [B,S,H*DH] f16
  constexpr int Sq = 2048, Dm = 1024, Dh = 64;
  const int lane  = threadIdx.x & 31;
  const int gw    = blockIdx.x * 8 + (threadIdx.x >> 5);   // 4096 waves total
  const int qtile = gw & 127;
  const int h     = (gw >> 7) & 15;
  const int b     = gw >> 11;
  const int l15   = lane & 15;
  const int g     = lane >> 4;
  const int q0    = qtile * 16;

  // Q^T B-fragments (k-dim = dh).  Fold the 1/sqrt(DH)=0.125 scale into Q.
  const _Float16* qptr = Q + (size_t)(b * Sq + q0 + l15) * Dm + h * Dh + g * 16;
  v16h bq0 = *(const v16h*)(qptr);        // dh 0..31
  v16h bq1 = *(const v16h*)(qptr + 32);   // dh 32..63
  const _Float16 sc = (_Float16)0.125f;
#pragma unroll
  for (int i = 0; i < 16; ++i) { bq0[i] *= sc; bq1[i] *= sc; }

  const _Float16* kbase = Km + (size_t)(b * Sq + l15) * Dm + h * Dh;
  const _Float16* vbase = Vt + (size_t)(b * Dm + h * Dh + l15) * Sq;

  v8f oacc[4];
#pragma unroll
  for (int i = 0; i < 4; ++i) oacc[i] = zero_v8f();
  float m_run = -3.0e38f, l_run = 0.f;

  for (int t0 = 0; t0 < Sq; t0 += 32) {
    // K A-fragments: [16 keys x 32 dh], rows contiguous in dh.
    v16h ak00 = load_afrag(kbase + (size_t)(t0)      * Dm,      g);
    v16h ak01 = load_afrag(kbase + (size_t)(t0)      * Dm + 32, g);
    v16h ak10 = load_afrag(kbase + (size_t)(t0 + 16) * Dm,      g);
    v16h ak11 = load_afrag(kbase + (size_t)(t0 + 16) * Dm + 32, g);

    v8f st0 = zero_v8f(), st1 = zero_v8f();   // S^T tiles: keys 0-15 / 16-31
    st0 = WMMA_F16(ak00, bq0, st0);
    st0 = WMMA_F16(ak01, bq1, st0);
    st1 = WMMA_F16(ak10, bq0, st1);
    st1 = WMMA_F16(ak11, bq1, st1);

    // Online softmax over keys (rows of S^T): lane holds query q=l15,
    // key rows r+8g; partner lane^16 holds the other 8 rows.
    float mloc = -3.0e38f;
#pragma unroll
    for (int r = 0; r < 8; ++r) mloc = fmaxf(mloc, fmaxf(st0[r], st1[r]));
    const float m_blk = fmaxf(mloc, __shfl_xor(mloc, 16, 32));
    const float m_new = fmaxf(m_run, m_blk);
    const float alpha = __expf(m_run - m_new);

    float p0[8], p1[8];
    float lsum = 0.f;
#pragma unroll
    for (int r = 0; r < 8; ++r) {
      p0[r] = __expf(st0[r] - m_new);
      p1[r] = __expf(st1[r] - m_new);
      lsum += p0[r] + p1[r];
    }
    l_run = l_run * alpha + lsum + __shfl_xor(lsum, 16, 32);
    m_run = m_new;

    // Rescale O accumulators.
#pragma unroll
    for (int dt = 0; dt < 4; ++dt)
#pragma unroll
      for (int i = 0; i < 8; ++i) oacc[dt][i] *= alpha;

    // Reshape P (C-layout) -> P^T B-fragment (k=keys, n=queries) via xor-16.
    v16h bp;
#pragma unroll
    for (int r = 0; r < 8; ++r) {
      const float s0 = __shfl_xor(p0[r], 16, 32);
      const float s1 = __shfl_xor(p1[r], 16, 32);
      const float lo = (g == 0) ? p0[r] : s1;   // k = g*16 + r
      const float hi = (g == 0) ? s0 : p1[r];   // k = g*16 + 8 + r
      bp[r]     = (_Float16)lo;
      bp[r + 8] = (_Float16)hi;
    }

    // O^T += V^T * P^T : A = V^T [16 dh x 32 keys], contiguous per lane.
#pragma unroll
    for (int dt = 0; dt < 4; ++dt) {
      v16h av = load_afrag(vbase + (size_t)(dt * 16) * Sq + t0, g);
      oacc[dt] = WMMA_F16(av, bp, oacc[dt]);
    }
  }

  const float linv = 1.0f / l_run;
  const size_t orow = (size_t)(b * Sq + q0 + l15) * Dm + h * Dh;
#pragma unroll
  for (int dt = 0; dt < 4; ++dt)
#pragma unroll
    for (int r = 0; r < 8; ++r) {
      const int dh = dt * 16 + r + g * 8;     // O^T C-layout row = dh
      O[orow + dh] = (_Float16)(oacc[dt][r] * linv);
    }
}

// ---------------------------------------------------------------------------
// Launch
// ---------------------------------------------------------------------------
extern "C" void kernel_launch(void* const* d_in, const int* in_sizes, int n_in,
                              void* d_out, int out_size, void* d_ws, size_t ws_size,
                              hipStream_t stream) {
  const float* x  = (const float*)d_in[0];
  const float* Wq = (const float*)d_in[1];
  const float* bq = (const float*)d_in[2];
  const float* Wk = (const float*)d_in[3];
  const float* bk = (const float*)d_in[4];
  const float* Wv = (const float*)d_in[5];
  const float* bv = (const float*)d_in[6];
  const float* Wo = (const float*)d_in[7];
  const float* bo = (const float*)d_in[8];

  constexpr size_t MB = 1u << 20;
  char* ws = (char*)d_ws;
  _Float16* X16  = (_Float16*)(ws);             //  8 MB  [B*S, D] f16 (reused as O16)
  _Float16* WqT  = (_Float16*)(ws +  8 * MB);   //  2 MB  [H*DH, D]
  _Float16* WkT  = (_Float16*)(ws + 10 * MB);   //  2 MB
  _Float16* WvT  = (_Float16*)(ws + 12 * MB);   //  2 MB
  _Float16* WoT  = (_Float16*)(ws + 14 * MB);   //  2 MB  [D, D] transposed
  _Float16* Q16  = (_Float16*)(ws + 16 * MB);   //  8 MB  [B,S,H*DH]
  _Float16* K16  = (_Float16*)(ws + 24 * MB);   //  8 MB  [B,S,H*DH]
  _Float16* Vt16 = (_Float16*)(ws + 32 * MB);   //  8 MB  [B,H,DH,S]
  _Float16* O16  = X16;                         // X16 dead after QKV GEMMs

  constexpr int M = 4096, N = 1024, K = 1024;

  // Stage 0: downconvert + repack (contiguous B-operand layout).
  cast_f32_to_f16_kernel<<<16384, 256, 0, stream>>>(x, X16, M * K);
  pack_w_qkv_kernel<<<4096, 256, 0, stream>>>(Wq, WqT);
  pack_w_qkv_kernel<<<4096, 256, 0, stream>>>(Wk, WkT);
  pack_w_qkv_kernel<<<4096, 256, 0, stream>>>(Wv, WvT);
  pack_wo_kernel<<<4096, 256, 0, stream>>>(Wo, WoT);

  // Stage 1: QKV projections.  (M/32)*(N/64) = 2048 waves -> 256 blocks.
  gemm_wmma_kernel<<<256, 256, 0, stream>>>(X16, WqT, bq, Q16, M, N, K, 0);
  gemm_wmma_kernel<<<256, 256, 0, stream>>>(X16, WkT, bk, K16, M, N, K, 0);
  gemm_wmma_kernel<<<256, 256, 0, stream>>>(X16, WvT, bv, Vt16, M, N, K, 1);

  // Stage 2: flash attention.  B*H*(S/16) = 4096 waves -> 512 blocks.
  flash_attn_kernel<<<512, 256, 0, stream>>>(Q16, K16, Vt16, O16);

  // Stage 3: output projection, fp32 result straight to d_out.
  gemm_wmma_kernel<<<256, 256, 0, stream>>>(O16, WoT, bo, (float*)d_out,
                                            M, N, K, 2);
}
